// GAThead3L32H12_56581899158200
// MI455X (gfx1250) — matrix operand order repaired and checked by reading, not compiled, not run on hardware
//
#include <hip/hip_runtime.h>
#include <hip/hip_bf16.h>

// ---------------------------------------------------------------------------
// Types for CDNA5 WMMA (wave32): D(16x16 f32) = A(16x32 bf16) * B(32x16 bf16) + C
// ---------------------------------------------------------------------------
typedef __attribute__((ext_vector_type(16))) __bf16 v16bf;
typedef __attribute__((ext_vector_type(8)))  float  v8f;

union FragBF {
    v16bf bf;
    uint4 q[2];
};

static __device__ __forceinline__ unsigned short f2bf(float f) {
    unsigned int u = __float_as_uint(f);
    unsigned int r = u + 0x7FFFu + ((u >> 16) & 1u);   // round-to-nearest-even
    return (unsigned short)(r >> 16);
}

// float -> order-preserving unsigned key (for atomicMax-based segment max)
static __device__ __forceinline__ unsigned int fkey(float f) {
    unsigned int u = __float_as_uint(f);
    return (u & 0x80000000u) ? ~u : (u | 0x80000000u);
}
static __device__ __forceinline__ float fdec(unsigned int k) {
    unsigned int u = (k & 0x80000000u) ? (k & 0x7FFFFFFFu) : ~k;
    return __uint_as_float(u);
}

// Generic pointer into LDS -> raw 32-bit LDS byte offset (ISA: addr[31:0]).
static __device__ __forceinline__ unsigned int lds_off(const void* p) {
    return (unsigned int)(unsigned long long)(uintptr_t)p;
}

// ---------------------------------------------------------------------------
// Weight packing: f32 W[K,N] row-major  ->  bf16 WMMA B fragments.
// Layout per 16-col tile ct, 32-row k-chunk kc, lane L (0..31), vgpr v (0..7):
//   uint32 = { B[kbase+2v][n] , B[kbase+2v+1][n] }  with
//   n = ct*16 + (L&15), kbase = kc*32 + (L>>4)*16.
// ---------------------------------------------------------------------------
__global__ void pack_w(const float* __restrict__ W, unsigned int* __restrict__ Bp,
                       int K, int N) {
    int idx = blockIdx.x * blockDim.x + threadIdx.x;
    int total = (K * N) >> 1;
    if (idx >= total) return;
    int v    = idx & 7;
    int lane = (idx >> 3) & 31;
    int rest = idx >> 8;
    int kcN  = K >> 5;
    int kc   = rest % kcN;
    int ct   = rest / kcN;
    int n    = ct * 16 + (lane & 15);
    int kb   = kc * 32 + (lane >> 4) * 16 + v * 2;
    unsigned int lo = f2bf(W[(size_t)kb * N + n]);
    unsigned int hi = f2bf(W[(size_t)(kb + 1) * N + n]);
    Bp[idx] = lo | (hi << 16);
}

__global__ void f32_to_bf16_k(const float* __restrict__ in,
                              unsigned short* __restrict__ out, size_t n) {
    size_t i = (size_t)blockIdx.x * blockDim.x + threadIdx.x;
    if (i < n) out[i] = f2bf(in[i]);
}

// ---------------------------------------------------------------------------
// bf16 WMMA GEMM: C[M,N] = A[M,K]*B[K,N] + bias[N]
// A: bf16 row-major. B: packed by pack_w. 256 threads = 8 waves, each wave
// one 16x16 tile of a 32x64 macro tile. Requires M%32==0, N%64==0, K%32==0.
// A tile is staged into LDS via the CDNA5 async Global->LDS engine
// (GLOBAL_LOAD_ASYNC_TO_LDS_B128, tracked with ASYNCcnt) -- no VGPR round-trip.
// ---------------------------------------------------------------------------
__global__ void __launch_bounds__(256)
gemm_bf16_wmma(const unsigned short* __restrict__ A,
               const unsigned int*  __restrict__ Bp,
               const float* __restrict__ bias,
               float* __restrict__ C, int M, int N, int K) {
    extern __shared__ unsigned short As[];          // 32 x K bf16 tile
    const int nCt4 = N >> 6;
    const int bm   = blockIdx.x / nCt4;
    const int bn   = blockIdx.x % nCt4;
    const int tid  = threadIdx.x;
    const int wave = tid >> 5;
    const int lane = tid & 31;

    // async stage of A tile into LDS (contiguous 32*K bf16 chunk, 16B units)
    {
        const unsigned long long gbase =
            (unsigned long long)(uintptr_t)(A + (size_t)bm * 32 * K);
        const unsigned int lbase = lds_off(&As[0]);
        const int n16 = (32 * K) >> 3;              // number of 16B chunks
        for (int i = tid; i < n16; i += 256) {
            unsigned long long ga = gbase + ((unsigned long long)i << 4);
            unsigned int       la = lbase + ((unsigned int)i << 4);
            asm volatile("global_load_async_to_lds_b128 %0, %1, off"
                         :: "v"(la), "v"(ga) : "memory");
        }
        asm volatile("s_wait_asynccnt 0" ::: "memory");
    }
    __syncthreads();

    const int rowTile = wave >> 2;                  // 0..1
    const int colTile = wave & 3;                   // 0..3
    const int ctG  = (bn << 2) + colTile;           // global 16-col tile
    const int m16  = lane & 15;
    const int half = lane >> 4;
    const int kcN  = K >> 5;

    const unsigned int* bptr = Bp + ((((size_t)ctG * kcN) * 32 + lane) << 3);
    __builtin_prefetch(bptr, 0, 1);

    v8f acc = {};
    for (int kc = 0; kc < kcN; ++kc) {
        FragBF a, b;
        // A fragment: lane holds row (rowTile*16+m16), K = kc*32 + half*16 .. +15
        // contiguous in v0..7.
        const unsigned short* arow =
            As + (size_t)(rowTile * 16 + m16) * K + kc * 32 + half * 16;
        a.q[0] = *(const uint4*)(arow);
        a.q[1] = *(const uint4*)(arow + 8);
        // B fragment: 8 contiguous uint32 per lane (pre-swizzled)
        const uint4* bq = (const uint4*)(bptr + ((size_t)kc << 8));
        b.q[0] = bq[0];
        b.q[1] = bq[1];
        acc = __builtin_amdgcn_wmma_f32_16x16x32_bf16(
            false, a.bf, false, b.bf, (short)0, acc, false, false);
    }

    // C tile: vgpr r -> row (base + r), lanes 16..31 cover rows +8
    const int   n0 = ctG * 16 + m16;
    const int   r0 = bm * 32 + rowTile * 16 + half * 8;
    const float bv = bias ? bias[n0] : 0.0f;
#pragma unroll
    for (int r = 0; r < 8; ++r)
        C[(size_t)(r0 + r) * N + n0] = acc[r] + bv;
}

// ---------------------------------------------------------------------------
// Edge phase kernels (layers 1-2, C=32 channels/head, one wave32 per edge)
// ---------------------------------------------------------------------------
__global__ void __launch_bounds__(256)
edge_logits(const float* __restrict__ xl, const float* __restrict__ xr,
            const float* __restrict__ ea, const float* __restrict__ We,
            const float* __restrict__ att, const long long* __restrict__ ei,
            float* __restrict__ lg, unsigned int* __restrict__ mx, int E, int H) {
    int gw   = (blockIdx.x * blockDim.x + threadIdx.x) >> 5;
    int lane = threadIdx.x & 31;
    if (gw >= E) return;
    long long s = ei[gw], d = ei[E + gw];
    const int HC = H * 32;
    float eav = (lane < 16) ? ea[(size_t)gw * 16 + lane] : 0.0f;
    for (int h = 0; h < H; ++h) {
        int hc = h * 32 + lane;
        float ep = 0.0f;
#pragma unroll
        for (int j = 0; j < 16; ++j)
            ep += __shfl(eav, j, 32) * We[(size_t)j * HC + hc];
        float v = xl[(size_t)s * HC + hc] + xr[(size_t)d * HC + hc] + ep;
        v = v > 0.0f ? v : 0.2f * v;                 // leaky relu
        float p = v * att[hc];
        for (int o = 16; o > 0; o >>= 1) p += __shfl_down(p, o, 32);
        if (lane == 0) {
            lg[(size_t)gw * H + h] = p;
            atomicMax(&mx[(size_t)d * H + h], fkey(p));
        }
    }
}

__global__ void edge_soft(float* __restrict__ lg, const unsigned int* __restrict__ mx,
                          float* __restrict__ den, const long long* __restrict__ ei,
                          int E, int H) {
    int i = blockIdx.x * blockDim.x + threadIdx.x;
    if (i >= E * H) return;
    int e = i / H, h = i - e * H;
    long long d = ei[E + e];
    float ex = __expf(lg[i] - fdec(mx[(size_t)d * H + h]));
    lg[i] = ex;                                      // in-place: logits -> exp
    atomicAdd(&den[(size_t)d * H + h], ex);
}

__global__ void __launch_bounds__(256)
edge_scatter(const float* __restrict__ xl, const float* __restrict__ ex,
             const float* __restrict__ den, const long long* __restrict__ ei,
             float* __restrict__ acc, int E, int H) {
    int gw   = (blockIdx.x * blockDim.x + threadIdx.x) >> 5;
    int lane = threadIdx.x & 31;
    if (gw >= E) return;
    long long s = ei[gw], d = ei[E + gw];
    const int HC = H * 32;
    for (int h = 0; h < H; ++h) {
        float a = ex[(size_t)gw * H + h] / (den[(size_t)d * H + h] + 1e-16f);
        int hc = h * 32 + lane;
        atomicAdd(&acc[(size_t)d * HC + hc], xl[(size_t)s * HC + hc] * a);
    }
}

// out = elu(acc + bias); emit f32 and/or bf16
__global__ void post_bias_elu(const float* __restrict__ acc, const float* __restrict__ bias,
                              unsigned short* __restrict__ obf, float* __restrict__ of32,
                              int HC, size_t total) {
    size_t i = (size_t)blockIdx.x * blockDim.x + threadIdx.x;
    if (i >= total) return;
    float v = acc[i] + bias[(int)(i % (size_t)HC)];
    float e = v > 0.0f ? v : __expf(v) - 1.0f;
    if (obf)  obf[i]  = f2bf(e);
    if (of32) of32[i] = e;
}

// ---------------------------------------------------------------------------
// Layer 3 (H=1, C=1): reductions
// ---------------------------------------------------------------------------
__global__ void __launch_bounds__(256)
dot_nodes(const float* __restrict__ Hin, const float* __restrict__ W,
          const float* __restrict__ b, float* __restrict__ out, int n, int K) {
    int gw   = (blockIdx.x * blockDim.x + threadIdx.x) >> 5;
    int lane = threadIdx.x & 31;
    if (gw >= n) return;
    float s = 0.0f;
    for (int j = lane; j < K; j += 32) s += Hin[(size_t)gw * K + j] * W[j];
    for (int o = 16; o > 0; o >>= 1) s += __shfl_down(s, o, 32);
    if (lane == 0) out[gw] = s + b[0];
}

__global__ void l3_logits(const float* __restrict__ xl3, const float* __restrict__ xr3,
                          const float* __restrict__ ea, const float* __restrict__ We3,
                          const float* __restrict__ att3, const long long* __restrict__ ei,
                          float* __restrict__ lg, unsigned int* __restrict__ mx, int E) {
    int e = blockIdx.x * blockDim.x + threadIdx.x;
    if (e >= E) return;
    long long s = ei[e], d = ei[E + e];
    float ep = 0.0f;
#pragma unroll
    for (int j = 0; j < 16; ++j) ep += ea[(size_t)e * 16 + j] * We3[j];
    float v = xl3[s] + xr3[d] + ep;
    v = v > 0.0f ? v : 0.2f * v;
    float l = v * att3[0];
    lg[e] = l;
    atomicMax(&mx[d], fkey(l));
}

__global__ void l3_soft(float* __restrict__ lg, const unsigned int* __restrict__ mx,
                        float* __restrict__ den, const long long* __restrict__ ei, int E) {
    int e = blockIdx.x * blockDim.x + threadIdx.x;
    if (e >= E) return;
    long long d = ei[E + e];
    float ex = __expf(lg[e] - fdec(mx[d]));
    lg[e] = ex;
    atomicAdd(&den[d], ex);
}

__global__ void l3_scatter(const float* __restrict__ xl3, const float* __restrict__ ex,
                           const float* __restrict__ den, const long long* __restrict__ ei,
                           float* __restrict__ acc, int E) {
    int e = blockIdx.x * blockDim.x + threadIdx.x;
    if (e >= E) return;
    long long s = ei[e], d = ei[E + e];
    float a = ex[e] / (den[d] + 1e-16f);
    atomicAdd(&acc[d], xl3[s] * a);
}

__global__ void l3_final(const float* __restrict__ acc, const float* __restrict__ b,
                         float* __restrict__ out, int n) {
    int i = blockIdx.x * blockDim.x + threadIdx.x;
    if (i < n) out[i] = acc[i] + b[0];
}

// ---------------------------------------------------------------------------
// Host driver
// ---------------------------------------------------------------------------
extern "C" void kernel_launch(void* const* d_in, const int* in_sizes, int n_in,
                              void* d_out, int out_size, void* d_ws, size_t ws_size,
                              hipStream_t stream) {
    (void)in_sizes; (void)n_in; (void)out_size; (void)ws_size;
    const int Nn = 60000, Ne = 360000;

    const float*      x    = (const float*)d_in[0];
    const long long*  ei   = (const long long*)d_in[1];
    const float*      ea   = (const float*)d_in[2];
    const float *Wl1 = (const float*)d_in[3],  *bl1 = (const float*)d_in[4];
    const float *Wr1 = (const float*)d_in[5],  *br1 = (const float*)d_in[6];
    const float *We1 = (const float*)d_in[7],  *att1 = (const float*)d_in[8];
    const float *bias1 = (const float*)d_in[9];
    const float *Wl2 = (const float*)d_in[10], *bl2 = (const float*)d_in[11];
    const float *Wr2 = (const float*)d_in[12], *br2 = (const float*)d_in[13];
    const float *We2 = (const float*)d_in[14], *att2 = (const float*)d_in[15];
    const float *bias2 = (const float*)d_in[16];
    const float *Wl3 = (const float*)d_in[17], *bl3 = (const float*)d_in[18];
    const float *Wr3 = (const float*)d_in[19], *br3 = (const float*)d_in[20];
    const float *We3 = (const float*)d_in[21], *att3 = (const float*)d_in[22];
    const float *bias3 = (const float*)d_in[23];
    float* out = (float*)d_out;

    // -------- workspace bump allocator (arena reuse across layers) --------
    char* base = (char*)d_ws;
    size_t off = 0;
    auto carve = [&](size_t bytes) -> void* {
        size_t o = (off + 255) & ~(size_t)255;
        off = o + bytes;
        return (void*)(base + o);
    };
    unsigned short* xbf  = (unsigned short*)carve((size_t)Nn * 64 * 2);
    unsigned int*   wl1p = (unsigned int*)carve((size_t)64 * 384 * 2);
    unsigned int*   wr1p = (unsigned int*)carve((size_t)64 * 384 * 2);
    unsigned int*   wl2p = (unsigned int*)carve((size_t)384 * 192 * 2);
    unsigned int*   wr2p = (unsigned int*)carve((size_t)384 * 192 * 2);
    float* region1 = (float*)carve((size_t)Nn * 384 * 4);   // xl1 -> acc2
    float* region2 = (float*)carve((size_t)Nn * 384 * 4);   // xr1 -> xl2
    float* region3 = (float*)carve((size_t)Nn * 384 * 4);   // acc1 -> xr2 + h2f
    unsigned short* h1bf = (unsigned short*)carve((size_t)Nn * 384 * 2);
    float*        ew  = (float*)carve((size_t)Ne * 12 * 4); // logits/exp (reused)
    unsigned int* mx  = (unsigned int*)carve((size_t)Nn * 12 * 4);
    float*        den = (float*)carve((size_t)Nn * 12 * 4);
    float* xl3  = (float*)carve((size_t)Nn * 4);
    float* xr3  = (float*)carve((size_t)Nn * 4);
    unsigned int* mx3 = (unsigned int*)carve((size_t)Nn * 4);
    float* den3 = (float*)carve((size_t)Nn * 4);
    float* acc3 = (float*)carve((size_t)Nn * 4);
    float* lg3  = (float*)carve((size_t)Ne * 4);

    float* xl1  = region1;
    float* xr1  = region2;
    float* acc1 = region3;
    float* xl2  = region2;                   // xr1 dead after layer-1 logits
    float* acc2 = region1;                   // xl1 dead after layer-1 scatter
    float* xr2  = region3;                   // acc1 dead after post1
    float* h2f  = region3 + (size_t)Nn * 192;

    // -------- pack weights + convert input activations to bf16 --------
    {
        int t1 = 64 * 384 / 2;
        pack_w<<<(t1 + 255) / 256, 256, 0, stream>>>(Wl1, wl1p, 64, 384);
        pack_w<<<(t1 + 255) / 256, 256, 0, stream>>>(Wr1, wr1p, 64, 384);
        int t2 = 384 * 192 / 2;
        pack_w<<<(t2 + 255) / 256, 256, 0, stream>>>(Wl2, wl2p, 384, 192);
        pack_w<<<(t2 + 255) / 256, 256, 0, stream>>>(Wr2, wr2p, 384, 192);
        size_t nx = (size_t)Nn * 64;
        f32_to_bf16_k<<<(unsigned)((nx + 255) / 256), 256, 0, stream>>>(x, xbf, nx);
    }

    const int ewBlocks  = (Ne + 7) / 8;      // wave-per-edge kernels
    // ======================= Layer 1 (H=12, HC=384) =======================
    {
        dim3 g((Nn / 32) * (384 / 64));
        gemm_bf16_wmma<<<g, 256, 32 * 64 * 2, stream>>>(xbf, wl1p, bl1, xl1, Nn, 384, 64);
        gemm_bf16_wmma<<<g, 256, 32 * 64 * 2, stream>>>(xbf, wr1p, br1, xr1, Nn, 384, 64);
        hipMemsetAsync(mx,  0, (size_t)Nn * 12 * 4, stream);
        hipMemsetAsync(den, 0, (size_t)Nn * 12 * 4, stream);
        hipMemsetAsync(acc1, 0, (size_t)Nn * 384 * 4, stream);
        edge_logits<<<ewBlocks, 256, 0, stream>>>(xl1, xr1, ea, We1, att1, ei, ew, mx, Ne, 12);
        edge_soft<<<(Ne * 12 + 255) / 256, 256, 0, stream>>>(ew, mx, den, ei, Ne, 12);
        edge_scatter<<<ewBlocks, 256, 0, stream>>>(xl1, ew, den, ei, acc1, Ne, 12);
        size_t tot = (size_t)Nn * 384;
        post_bias_elu<<<(unsigned)((tot + 255) / 256), 256, 0, stream>>>(
            acc1, bias1, h1bf, (float*)nullptr, 384, tot);
    }
    // ======================= Layer 2 (H=6, HC=192) ========================
    {
        dim3 g((Nn / 32) * (192 / 64));
        gemm_bf16_wmma<<<g, 256, 32 * 384 * 2, stream>>>(h1bf, wl2p, bl2, xl2, Nn, 192, 384);
        gemm_bf16_wmma<<<g, 256, 32 * 384 * 2, stream>>>(h1bf, wr2p, br2, xr2, Nn, 192, 384);
        hipMemsetAsync(mx,  0, (size_t)Nn * 6 * 4, stream);
        hipMemsetAsync(den, 0, (size_t)Nn * 6 * 4, stream);
        hipMemsetAsync(acc2, 0, (size_t)Nn * 192 * 4, stream);
        edge_logits<<<ewBlocks, 256, 0, stream>>>(xl2, xr2, ea, We2, att2, ei, ew, mx, Ne, 6);
        edge_soft<<<(Ne * 6 + 255) / 256, 256, 0, stream>>>(ew, mx, den, ei, Ne, 6);
        edge_scatter<<<ewBlocks, 256, 0, stream>>>(xl2, ew, den, ei, acc2, Ne, 6);
        size_t tot = (size_t)Nn * 192;
        post_bias_elu<<<(unsigned)((tot + 255) / 256), 256, 0, stream>>>(
            acc2, bias2, (unsigned short*)nullptr, h2f, 192, tot);
    }
    // ======================= Layer 3 (H=1, C=1) ===========================
    {
        int nb = (Nn + 7) / 8;
        dot_nodes<<<nb, 256, 0, stream>>>(h2f, Wl3, bl3, xl3, Nn, 192);
        dot_nodes<<<nb, 256, 0, stream>>>(h2f, Wr3, br3, xr3, Nn, 192);
        hipMemsetAsync(mx3,  0, (size_t)Nn * 4, stream);
        hipMemsetAsync(den3, 0, (size_t)Nn * 4, stream);
        hipMemsetAsync(acc3, 0, (size_t)Nn * 4, stream);
        int eb = (Ne + 255) / 256;
        l3_logits<<<eb, 256, 0, stream>>>(xl3, xr3, ea, We3, att3, ei, lg3, mx3, Ne);
        l3_soft<<<eb, 256, 0, stream>>>(lg3, mx3, den3, ei, Ne);
        l3_scatter<<<eb, 256, 0, stream>>>(xl3, lg3, den3, ei, acc3, Ne);
        l3_final<<<(Nn + 255) / 256, 256, 0, stream>>>(acc3, bias3, out, Nn);
    }
}
